// BaseRelationNetwork_20074677141789
// MI455X (gfx1250) — matrix operand order repaired and checked by reading, not compiled, not run on hardware
//
#include <hip/hip_runtime.h>

// ---------------------------------------------------------------------------
// BaseRelationNetwork for MI455X (gfx1250, wave32, WMMA f32_16x16x32_f16)
//
// out[b] = mean_p ( relu(relu([x_i|x_j]@W1+b1)@W2+b2) @ W3 + b3 )
//
// Optimizations:
//  1) Layer-1 split:  [x_i|x_j]@W1 = x_i@W1_top + x_j@W1_bot  -> per-channel GEMM
//     (8.6 GFLOP instead of 64)
//  2) mean-before-W3: mean_p(h2)@W3 + b3   (layer 3 shrinks 120x)
//  Precision: f16 inputs to WMMA, fp32 accumulation everywhere.
//
//  gemm1 tiling: one wave owns a 32x64 output tile -> 2 A-frags x 4 B-frags
//  feed 8 WMMAs per K-step (1 KB of fragment traffic per WMMA).
// ---------------------------------------------------------------------------

typedef __attribute__((ext_vector_type(16))) _Float16 v16h;
typedef __attribute__((ext_vector_type(8)))  _Float16 v8h;
typedef __attribute__((ext_vector_type(8)))  float    v8f;
typedef __attribute__((ext_vector_type(4)))  float    v4f;

#define NB   512          // batch
#define NC   16           // channels
#define NF   1024         // input features
#define NH   256          // hidden
#define NP   120          // pairs = NC*(NC-1)/2
#define M1   (NB * NC)    // 8192 rows for GEMM1
#define N1   (2 * NH)     // 512 cols: [0,256)=A (x@W1_top), [256,512)=Bm (x@W1_bot)

// ---- WMMA fragment loaders -------------------------------------------------
// A-fragment (16-bit A, 16x32): lanes 0-15: M=lane, K={k0..k0+7, k0+16..k0+23}
//                               lanes 16-31: M=lane-16, K offset by +8.
__device__ __forceinline__ v16h load_a_frag_f16(const _Float16* __restrict__ base,
                                                int stride, int m0, int k0, int lane) {
    const int m  = m0 + (lane & 15);
    const int kk = k0 + ((lane & 16) ? 8 : 0);
    const _Float16* p = base + (size_t)m * stride + kk;
    v8h lo = *(const v8h*)(p);
    v8h hi = *(const v8h*)(p + 16);
    v16h a;
#pragma unroll
    for (int i = 0; i < 8; ++i) { a[i] = lo[i]; a[i + 8] = hi[i]; }
    return a;
}

// B-fragment (16-bit B, 32x16) from transposed weights W^T stored [N][K]:
// lane gives column N = n0 + (lane&15); K = {kk..kk+7, kk+16..kk+23}, kk = k0 + (lane>=16 ? 8 : 0).
__device__ __forceinline__ v16h load_b_frag(const _Float16* __restrict__ wt,
                                            int stride, int n0, int k0, int lane) {
    const int n  = n0 + (lane & 15);
    const int kk = k0 + ((lane & 16) ? 8 : 0);
    const _Float16* p = wt + (size_t)n * stride + kk;
    v8h lo = *(const v8h*)(p);
    v8h hi = *(const v8h*)(p + 16);
    v16h b;
#pragma unroll
    for (int i = 0; i < 8; ++i) { b[i] = lo[i]; b[i + 8] = hi[i]; }
    return b;
}

// ---- Weight conversion / transpose kernels --------------------------------
// W1 [2F,H] -> w1t f16 [512][1024]: row n<256 is W1_top column n, n>=256 is W1_bot.
__global__ void cvt_w1t_kernel(const float* __restrict__ w1, _Float16* __restrict__ w1t) {
    const int idx = blockIdx.x * blockDim.x + threadIdx.x;   // 0..524287
    const int n = idx >> 10;
    const int k = idx & 1023;
    const float v = (n < NH) ? w1[(size_t)k * NH + n]
                             : w1[(size_t)(k + NF) * NH + (n - NH)];
    w1t[idx] = (_Float16)v;
}

// W2,W3 [H,H] -> f16 transposed [N][K]
__global__ void cvt_w23t_kernel(const float* __restrict__ w2, const float* __restrict__ w3,
                                _Float16* __restrict__ w2t, _Float16* __restrict__ w3t) {
    const int idx = blockIdx.x * blockDim.x + threadIdx.x;   // 0..131071
    const int sel = idx >> 16;
    const int r = idx & 65535;
    const int n = r >> 8, k = r & 255;
    const float* src = sel ? w3 : w2;
    _Float16*   dst = sel ? w3t : w2t;
    dst[(size_t)n * NH + k] = (_Float16)src[(size_t)k * NH + n];
}

// ---- GEMM1: AB[8192,512] = x[8192,1024] @ [W1_top | W1_bot] ----------------
// One wave per 32x64 output tile: 2 A-frags + 4 B-frags -> 8 WMMAs per K-step.
__global__ __launch_bounds__(128) void gemm1_kernel(const float* __restrict__ x,
                                                    const _Float16* __restrict__ w1t,
                                                    float* __restrict__ ab) {
    const int wid  = blockIdx.x * 4 + (threadIdx.x >> 5);    // 0..2047
    const int lane = threadIdx.x & 31;
    const int m0 = (wid >> 3) * 32;                          // 256 M-tiles of 32 rows
    const int n0 = (wid & 7) * 64;                           // 8 N-strips of 64 cols
    v8f acc[2][4] = {};
    for (int k0 = 0; k0 < NF; k0 += 32) {
        // Two A fragments (rows m0..m0+15 and m0+16..m0+31), fp32 -> f16 in-register
        v16h a[2];
#pragma unroll
        for (int h = 0; h < 2; ++h) {
            const int m  = m0 + h * 16 + (lane & 15);
            const int kk = k0 + ((lane & 16) ? 8 : 0);
            const float* p = x + (size_t)m * NF + kk;
            v4f a0 = *(const v4f*)(p);
            v4f a1 = *(const v4f*)(p + 4);
            v4f a2 = *(const v4f*)(p + 16);
            v4f a3 = *(const v4f*)(p + 20);
#pragma unroll
            for (int i = 0; i < 4; ++i) {
                a[h][i]      = (_Float16)a0[i];
                a[h][i + 4]  = (_Float16)a1[i];
                a[h][i + 8]  = (_Float16)a2[i];
                a[h][i + 12] = (_Float16)a3[i];
            }
        }
#pragma unroll
        for (int t = 0; t < 4; ++t) {
            v16h bfrag = load_b_frag(w1t, NF, n0 + t * 16, k0, lane);
#pragma unroll
            for (int h = 0; h < 2; ++h) {
                acc[h][t] = __builtin_amdgcn_wmma_f32_16x16x32_f16(
                    false, a[h], false, bfrag, (short)0, acc[h][t], false, false);
            }
        }
    }
    const int nlo = lane & 15;
#pragma unroll
    for (int h = 0; h < 2; ++h) {
        const int mrow = m0 + h * 16 + ((lane & 16) ? 8 : 0);
#pragma unroll
        for (int t = 0; t < 4; ++t) {
            float* o = ab + (size_t)mrow * N1 + n0 + t * 16 + nlo;
#pragma unroll
            for (int r = 0; r < 8; ++r) o[(size_t)r * N1] = acc[h][t][r];
        }
    }
}

// ---- Pair kernel: per batch b, all 120 pairs through layer 2, sum over p ---
// 16 waves = 16 N-tiles of W2; 8 groups of 16 pairs; h1 tile staged in LDS.
__global__ __launch_bounds__(512) void pair_kernel(const float* __restrict__ ab,
                                                   const float* __restrict__ b1,
                                                   const _Float16* __restrict__ w2t,
                                                   const float* __restrict__ b2,
                                                   float* __restrict__ ssum) {
    __shared__ _Float16 h1[16 * NH];   // 8 KB: 16 pair-rows x 256 K, f16
    __shared__ float    acc_lds[NH];   // per-column sum over pairs

    const int b    = blockIdx.x;
    const int tid  = threadIdx.x;
    const int wave = tid >> 5;         // 0..15 -> N-tile
    const int lane = tid & 31;

    if (tid < NH) acc_lds[tid] = 0.0f;

    // Preload this wave's W2^T B-fragments for all 8 K-steps (reused by all groups)
    v16h bf[8];
#pragma unroll
    for (int k = 0; k < 8; ++k) bf[k] = load_b_frag(w2t, NH, wave * 16, k * 32, lane);

    const int   ncol  = wave * 16 + (lane & 15);
    const float b2n   = b2[ncol];
    const int   rbase = (lane & 16) ? 8 : 0;
    float lanesum = 0.0f;

    const int r_build = tid >> 5;          // row of h1 this thread builds
    const int kb      = lane * 8;          // 8-wide K chunk

    for (int g = 0; g < 8; ++g) {
        __syncthreads();                   // previous group's h1 reads done
        {   // build h1[r][k] = relu(A[b][i] + Bm[b][j] + b1) in f16
            const int p = g * 16 + r_build;
            _Float16* dst = &h1[r_build * NH + kb];
            if (p < NP) {
                int i = 0, base = 0;                       // triu_indices order
                while (p >= base + (15 - i)) { base += 15 - i; ++i; }
                const int j = p - base + i + 1;
                const float* pa  = ab + (size_t)(b * NC + i) * N1 + kb;       // A part
                const float* pbm = ab + (size_t)(b * NC + j) * N1 + NH + kb;  // Bm part
                const float* pb1 = b1 + kb;
#pragma unroll
                for (int q = 0; q < 8; ++q) {
                    float v = pa[q] + pbm[q] + pb1[q];
                    dst[q] = (_Float16)(v > 0.0f ? v : 0.0f);
                }
            } else {
#pragma unroll
                for (int q = 0; q < 8; ++q) dst[q] = (_Float16)0.0f;
            }
        }
        __syncthreads();                   // h1 tile visible to all waves
        v8f acc = {};
#pragma unroll
        for (int k = 0; k < 8; ++k) {
            v16h a = load_a_frag_f16(h1, NH, 0, k * 32, lane);   // from LDS
            acc = __builtin_amdgcn_wmma_f32_16x16x32_f16(
                false, a, false, bf[k], (short)0, acc, false, false);
        }
        // h2 = relu(acc + b2); sum over valid pair rows (mask dummies!)
#pragma unroll
        for (int r = 0; r < 8; ++r) {
            const int p = g * 16 + rbase + r;
            float v = acc[r] + b2n;
            v = v > 0.0f ? v : 0.0f;
            lanesum += (p < NP) ? v : 0.0f;
        }
    }
    atomicAdd(&acc_lds[ncol], lanesum);    // ds_add_f32; 2 lanes per column
    __syncthreads();
    if (tid < NH) ssum[(size_t)b * NH + tid] = acc_lds[tid];
}

// ---- GEMM3: out[512,256] = (Ssum/120) @ W3 + b3 ----------------------------
__global__ __launch_bounds__(128) void gemm3_kernel(const float* __restrict__ ssum,
                                                    const _Float16* __restrict__ w3t,
                                                    const float* __restrict__ b3,
                                                    float* __restrict__ out) {
    const int wid  = blockIdx.x * 4 + (threadIdx.x >> 5);    // 0..511
    const int lane = threadIdx.x & 31;
    const int m0 = (wid >> 4) * 16;
    const int n0 = (wid & 15) * 16;
    const float invP = 1.0f / (float)NP;
    v8f acc = {};
    for (int k0 = 0; k0 < NH; k0 += 32) {
        const int m  = m0 + (lane & 15);
        const int kk = k0 + ((lane & 16) ? 8 : 0);
        const float* p = ssum + (size_t)m * NH + kk;
        v4f a0 = *(const v4f*)(p);
        v4f a1 = *(const v4f*)(p + 4);
        v4f a2 = *(const v4f*)(p + 16);
        v4f a3 = *(const v4f*)(p + 20);
        v16h a;
#pragma unroll
        for (int i = 0; i < 4; ++i) {
            a[i]      = (_Float16)(a0[i] * invP);
            a[i + 4]  = (_Float16)(a1[i] * invP);
            a[i + 8]  = (_Float16)(a2[i] * invP);
            a[i + 12] = (_Float16)(a3[i] * invP);
        }
        v16h bfrag = load_b_frag(w3t, NH, n0, k0, lane);
        acc = __builtin_amdgcn_wmma_f32_16x16x32_f16(
            false, a, false, bfrag, (short)0, acc, false, false);
    }
    const int n    = n0 + (lane & 15);
    const int mrow = m0 + ((lane & 16) ? 8 : 0);
    const float bias = b3[n];
#pragma unroll
    for (int r = 0; r < 8; ++r) out[(size_t)(mrow + r) * NH + n] = acc[r] + bias;
}

// ---------------------------------------------------------------------------
extern "C" void kernel_launch(void* const* d_in, const int* in_sizes, int n_in,
                              void* d_out, int out_size, void* d_ws, size_t ws_size,
                              hipStream_t stream) {
    const float* x  = (const float*)d_in[0];
    const float* W1 = (const float*)d_in[1];
    const float* b1 = (const float*)d_in[2];
    const float* W2 = (const float*)d_in[3];
    const float* b2 = (const float*)d_in[4];
    const float* W3 = (const float*)d_in[5];
    const float* b3 = (const float*)d_in[6];
    float* out = (float*)d_out;

    // Workspace layout (~18.5 MB total)
    char* ws = (char*)d_ws;
    _Float16* w1t  = (_Float16*)(ws);                                  // 1 MiB
    _Float16* w2t  = (_Float16*)(ws + (1u << 20));                     // 128 KiB
    _Float16* w3t  = (_Float16*)(ws + (1u << 20) + (128u << 10));      // 128 KiB
    float*    ab   = (float*)   (ws + (2u << 20));                     // 16 MiB
    float*    ssum = (float*)   (ws + (18u << 20));                    // 512 KiB

    cvt_w1t_kernel <<<2048, 256, 0, stream>>>(W1, w1t);
    cvt_w23t_kernel<<< 512, 256, 0, stream>>>(W2, W3, w2t, w3t);
    gemm1_kernel   <<< 512, 128, 0, stream>>>(x, w1t, ab);
    pair_kernel    <<< 512, 512, 0, stream>>>(ab, b1, w2t, b2, ssum);
    gemm3_kernel   <<< 128, 128, 0, stream>>>(ssum, w3t, b3, out);
}
